// AlbertBiGruCrf_79946521248239
// MI455X (gfx1250) — matrix operand order repaired and checked by reading, not compiled, not run on hardware
//
#include <hip/hip_runtime.h>
#include <hip/hip_bf16.h>
#include <math.h>

typedef __attribute__((ext_vector_type(16))) __bf16 v16bf;
typedef __attribute__((ext_vector_type(8)))  float  v8f;
typedef __attribute__((ext_vector_type(4)))  unsigned int v4u;
typedef __attribute__((ext_vector_type(8)))  int v8i;
typedef __attribute__((ext_vector_type(4)))  int v4i;

#define B_   32
#define S_   256
#define H_   768
#define E_   128
#define NH_  12
#define DH_  64
#define FFN_ 3072
#define G_   128
#define L_   9
#define NL_  12

#if defined(__has_builtin)
#  if __has_builtin(__builtin_amdgcn_tensor_load_to_lds) && __has_builtin(__builtin_amdgcn_s_wait_tensorcnt)
#    define HAVE_TDM 1
#  else
#    define HAVE_TDM 0
#  endif
#  if __has_builtin(__builtin_amdgcn_update_dpp)
#    define HAVE_DPP 1
#  else
#    define HAVE_DPP 0
#  endif
#else
#  define HAVE_TDM 0
#  define HAVE_DPP 0
#endif

// ---- CDNA5 WMMA helpers -----------------------------------------------------
// A-matrix 16x32 bf16 per-lane K index (ISA 7.12.2).
__device__ __forceinline__ int a_kidx(int v, int h, int lh) {
  return ((v >> 2) << 4) | (lh << 3) | ((v & 3) << 1) | h;
}
// B-matrix 32x16: same K order as A (lanes 0-15 K=0-7,16-23; lanes 16-31 K=8-15,24-31).
__device__ __forceinline__ int b_kidx(int v, int h, int lh) {
  return ((v >> 2) << 4) | (lh << 3) | ((v & 3) << 1) | h;
}

__device__ __forceinline__ v8f wmma_bf16(v16bf a, v16bf b, v8f c) {
  return __builtin_amdgcn_wmma_f32_16x16x32_bf16(
      false, a, false, b, (short)0, c, false, false);
}

// ---- wave32 16-lane XOR-tree reductions (VALU/DPP, no LDS traffic) ----------
__device__ __forceinline__ float xor16_max(float x) {
#if HAVE_DPP
  int t;
  t = __builtin_amdgcn_update_dpp(0, __float_as_int(x), 0xB1, 0xF, 0xF, true);   // quad_perm xor1
  x = fmaxf(x, __int_as_float(t));
  t = __builtin_amdgcn_update_dpp(0, __float_as_int(x), 0x4E, 0xF, 0xF, true);   // quad_perm xor2
  x = fmaxf(x, __int_as_float(t));
  t = __builtin_amdgcn_update_dpp(0, __float_as_int(x), 0x141, 0xF, 0xF, true);  // row_half_mirror
  x = fmaxf(x, __int_as_float(t));
  t = __builtin_amdgcn_update_dpp(0, __float_as_int(x), 0x140, 0xF, 0xF, true);  // row_mirror
  x = fmaxf(x, __int_as_float(t));
#else
  for (int d = 1; d < 16; d <<= 1) x = fmaxf(x, __shfl_xor(x, d, 32));
#endif
  return x;
}
__device__ __forceinline__ float xor16_sum(float x) {
#if HAVE_DPP
  int t;
  t = __builtin_amdgcn_update_dpp(0, __float_as_int(x), 0xB1, 0xF, 0xF, true);
  x += __int_as_float(t);
  t = __builtin_amdgcn_update_dpp(0, __float_as_int(x), 0x4E, 0xF, 0xF, true);
  x += __int_as_float(t);
  t = __builtin_amdgcn_update_dpp(0, __float_as_int(x), 0x141, 0xF, 0xF, true);
  x += __int_as_float(t);
  t = __builtin_amdgcn_update_dpp(0, __float_as_int(x), 0x140, 0xF, 0xF, true);
  x += __int_as_float(t);
#else
  for (int d = 1; d < 16; d <<= 1) x += __shfl_xor(x, d, 32);
#endif
  return x;
}

// ---- Tensor Data Mover: DMA a 2D fp32 tile (tile_x cols, tile_y rows, row
// stride in elements) from global into LDS, optional 1-DWORD pad per 32 DWORDs.
#if HAVE_TDM
__device__ __forceinline__ void tdm_load_2d(
    unsigned lds_off, const float* gptr, unsigned tile_x, unsigned tile_y,
    unsigned long long row_stride_elems, bool pad)
{
  unsigned long long ga = (unsigned long long)(size_t)gptr;
  v4u g0;
  g0[0] = 1u;                                          // count=1 (user mode)
  g0[1] = lds_off;                                     // lds_addr (bytes)
  g0[2] = (unsigned)(ga & 0xffffffffull);              // global_addr[31:0]
  g0[3] = (unsigned)((ga >> 32) & 0x01fffffful) | (2u << 30);  // addr[56:32] | type=2
  v8i g1;
  unsigned w0 = (2u << 16);                            // data_size = 4B
  if (pad) w0 |= (1u << 20) | (4u << 22) | (0u << 25); // pad: 1 DWORD per 32 DWORDs
  g1[0] = (int)w0;
  g1[1] = (int)((tile_x & 0xffffu) << 16);                               // tensor_dim0 lo16
  g1[2] = (int)(((tile_x >> 16) & 0xffffu) | ((tile_y & 0xffffu) << 16));// dim0 hi | dim1 lo
  g1[3] = (int)(((tile_y >> 16) & 0xffffu) | (tile_x << 16));            // dim1 hi | tile_dim0
  g1[4] = (int)(tile_y & 0xffffu);                                       // tile_dim1 (tile_dim2=0)
  g1[5] = (int)(row_stride_elems & 0xffffffffull);                       // dim0_stride lo32
  g1[6] = (int)((row_stride_elems >> 32) & 0xffffull);                   // dim0_stride hi16
  g1[7] = 0;
  v4i z4 = {0, 0, 0, 0};
  v8i z8 = {0, 0, 0, 0, 0, 0, 0, 0};
  __builtin_amdgcn_tensor_load_to_lds(g0, g1, z4, z4, z8, 0);
}
#endif

// ---- Generic WMMA GEMM: C[M,N] = act(A[M,K] x B + bias) --------------------
// TRANSB==0: B is [K,N] row-major. TRANSB==1: B is [N,K] row-major (A x B^T).
// ACT: 0 = none, 1 = tanh-GELU. Requires M%128==0, N%128==0, K%32==0.
// fp32 tiles staged in LDS (via TDM when available), bf16 cvt at fragment build.
template<int ACT, int TRANSB>
__global__ __launch_bounds__(256) void gemm_kernel(
    const float* __restrict__ A, const float* __restrict__ Bm,
    const float* __restrict__ bias, float* __restrict__ C,
    int M, int N, int K)
{
  __shared__ float As[128 * 33];                 // 128 rows x 32 (+1 pad)
  __shared__ float Bs[128 * 33];                 // TRANSB: 128x(32+1); else 32x128
  const int ASTRIDE = 33;
  const int BSTRIDE = TRANSB ? 33 : 128;
  const int tid  = threadIdx.x;
  const int lane = tid & 31, wid = tid >> 5;
  const int lm = lane & 15, lh = lane >> 4;
  const int wr = wid >> 2, wc = wid & 3;         // 2x4 wave grid
  const int row0 = blockIdx.y * 128;
  const int col0 = blockIdx.x * 128;

  const v8f vzero = {0, 0, 0, 0, 0, 0, 0, 0};
  v8f acc[4][2];
  #pragma unroll
  for (int i = 0; i < 4; i++)
    #pragma unroll
    for (int j = 0; j < 2; j++) acc[i][j] = vzero;

#if HAVE_TDM
  const unsigned ldsA = (unsigned)(size_t)(void*)As;
  const unsigned ldsB = (unsigned)(size_t)(void*)Bs;
#endif

  for (int k0 = 0; k0 < K; k0 += 32) {
#if HAVE_TDM
    if (wid == 0) {                              // one wave drives the TDM
      tdm_load_2d(ldsA, A + (size_t)row0 * K + k0, 32u, 128u,
                  (unsigned long long)K, true);
      if (TRANSB)
        tdm_load_2d(ldsB, Bm + (size_t)col0 * K + k0, 32u, 128u,
                    (unsigned long long)K, true);
      else
        tdm_load_2d(ldsB, Bm + (size_t)k0 * N + col0, 128u, 32u,
                    (unsigned long long)N, false);
      __builtin_amdgcn_s_wait_tensorcnt(0);
    }
    __syncthreads();
#else
    #pragma unroll
    for (int i = 0; i < 16; i++) {               // 128x32 A tile
      int idx = tid + 256 * i;
      int r = idx >> 5, c = idx & 31;
      As[r * ASTRIDE + c] = A[(size_t)(row0 + r) * K + k0 + c];
    }
    #pragma unroll
    for (int i = 0; i < 16; i++) {               // B tile
      int idx = tid + 256 * i;
      if (TRANSB) {
        int r = idx >> 5, c = idx & 31;          // [n][k]
        Bs[r * BSTRIDE + c] = Bm[(size_t)(col0 + r) * K + k0 + c];
      } else {
        int r = idx >> 7, c = idx & 127;         // [k][n]
        Bs[r * BSTRIDE + c] = Bm[(size_t)(k0 + r) * N + col0 + c];
      }
    }
    __syncthreads();
#endif

    v16bf bfrag[2];
    #pragma unroll
    for (int nt = 0; nt < 2; nt++) {
      int bn = wc * 32 + nt * 16 + lm;
      #pragma unroll
      for (int v = 0; v < 8; v++) {
        if (TRANSB) {
          bfrag[nt][2*v+0] = (__bf16)Bs[bn * BSTRIDE + b_kidx(v, 0, lh)];
          bfrag[nt][2*v+1] = (__bf16)Bs[bn * BSTRIDE + b_kidx(v, 1, lh)];
        } else {
          bfrag[nt][2*v+0] = (__bf16)Bs[b_kidx(v, 0, lh) * BSTRIDE + bn];
          bfrag[nt][2*v+1] = (__bf16)Bs[b_kidx(v, 1, lh) * BSTRIDE + bn];
        }
      }
    }
    #pragma unroll
    for (int mt = 0; mt < 4; mt++) {
      int am = wr * 64 + mt * 16 + lm;
      v16bf afrag;
      #pragma unroll
      for (int v = 0; v < 8; v++) {
        afrag[2*v+0] = (__bf16)As[am * ASTRIDE + a_kidx(v, 0, lh)];
        afrag[2*v+1] = (__bf16)As[am * ASTRIDE + a_kidx(v, 1, lh)];
      }
      #pragma unroll
      for (int nt = 0; nt < 2; nt++)
        acc[mt][nt] = wmma_bf16(afrag, bfrag[nt], acc[mt][nt]);
    }
    __syncthreads();
  }

  #pragma unroll
  for (int mt = 0; mt < 4; mt++) {
    #pragma unroll
    for (int nt = 0; nt < 2; nt++) {
      int col = col0 + wc * 32 + nt * 16 + lm;
      float bv = bias ? bias[col] : 0.0f;
      #pragma unroll
      for (int r = 0; r < 8; r++) {
        int row = row0 + wr * 64 + mt * 16 + r + 8 * lh;
        float x = acc[mt][nt][r] + bv;
        if (ACT == 1) {
          float x3 = x * x * x;
          x = 0.5f * x * (1.0f + tanhf(0.7978845608028654f * (x + 0.044715f * x3)));
        }
        C[(size_t)row * N + col] = x;
      }
    }
  }
}

// ---- Fused flash attention per (batch, head) -------------------------------
__global__ __launch_bounds__(256) void attn_kernel(
    const float* __restrict__ Q, const float* __restrict__ Kx,
    const float* __restrict__ Vx, const int* __restrict__ mask,
    float* __restrict__ Ctx)
{
  __shared__ __bf16 Ksm[S_][DH_];
  __shared__ __bf16 Vsm[S_][DH_];
  __shared__ __bf16 Psm[S_][32];
  __shared__ float  biasm[S_];
  const int bh = blockIdx.x;
  const int b = bh / NH_, hd = bh % NH_;
  const int tid  = threadIdx.x;
  const int lane = tid & 31, wid = tid >> 5;
  const int lm = lane & 15, lh = lane >> 4;
  const float* Qp = Q  + (size_t)b * S_ * H_ + hd * DH_;
  const float* Kp = Kx + (size_t)b * S_ * H_ + hd * DH_;
  const float* Vp = Vx + (size_t)b * S_ * H_ + hd * DH_;

  #pragma unroll 8
  for (int i = 0; i < 64; i++) {
    int idx = tid + 256 * i;
    int r = idx >> 6, c = idx & 63;
    Ksm[r][c] = (__bf16)Kp[(size_t)r * H_ + c];
    Vsm[r][c] = (__bf16)Vp[(size_t)r * H_ + c];
  }
  biasm[tid] = (1.0f - (float)mask[b * S_ + tid]) * -1e4f;
  __syncthreads();

  v16bf qf[2][2];                                // rows 32*wid..32*wid+31
  #pragma unroll
  for (int mt = 0; mt < 2; mt++) {
    int qrow = 32 * wid + 16 * mt + lm;
    #pragma unroll
    for (int ks = 0; ks < 2; ks++)
      #pragma unroll
      for (int v = 0; v < 8; v++) {
        qf[mt][ks][2*v+0] = (__bf16)Qp[(size_t)qrow * H_ + 32 * ks + a_kidx(v, 0, lh)];
        qf[mt][ks][2*v+1] = (__bf16)Qp[(size_t)qrow * H_ + 32 * ks + a_kidx(v, 1, lh)];
      }
  }

  const v8f vzero = {0, 0, 0, 0, 0, 0, 0, 0};
  float mxv[2][8], smv[2][8];
  v8f o[2][4];
  #pragma unroll
  for (int mt = 0; mt < 2; mt++) {
    #pragma unroll
    for (int r = 0; r < 8; r++) { mxv[mt][r] = -1e30f; smv[mt][r] = 0.0f; }
    #pragma unroll
    for (int nt = 0; nt < 4; nt++) o[mt][nt] = vzero;
  }

  for (int c = 0; c < S_ / 32; c++) {            // 32-key chunks
    v8f st[2][2] = {{vzero, vzero}, {vzero, vzero}};
    #pragma unroll
    for (int ks = 0; ks < 2; ks++) {
      v16bf kf[2];
      #pragma unroll
      for (int nt = 0; nt < 2; nt++) {
        int key = 32 * c + 16 * nt + lm;
        #pragma unroll
        for (int v = 0; v < 8; v++) {
          kf[nt][2*v+0] = Ksm[key][32 * ks + b_kidx(v, 0, lh)];
          kf[nt][2*v+1] = Ksm[key][32 * ks + b_kidx(v, 1, lh)];
        }
      }
      #pragma unroll
      for (int mt = 0; mt < 2; mt++)
        #pragma unroll
        for (int nt = 0; nt < 2; nt++)
          st[mt][nt] = wmma_bf16(qf[mt][ks], kf[nt], st[mt][nt]);
    }
    // online softmax (rows = r + 8*lh; 16 columns striped over lanes)
    #pragma unroll
    for (int mt = 0; mt < 2; mt++) {
      #pragma unroll
      for (int r = 0; r < 8; r++) {
        float s0 = st[mt][0][r] * 0.125f + biasm[32 * c + lm];
        float s1 = st[mt][1][r] * 0.125f + biasm[32 * c + 16 + lm];
        float cm = xor16_max(fmaxf(s0, s1));
        float nm = fmaxf(mxv[mt][r], cm);
        float corr = __expf(mxv[mt][r] - nm);
        mxv[mt][r] = nm;
        smv[mt][r] *= corr;
        #pragma unroll
        for (int nt = 0; nt < 4; nt++) o[mt][nt][r] *= corr;
        float p0 = __expf(s0 - nm), p1 = __expf(s1 - nm);
        smv[mt][r] += xor16_sum(p0 + p1);
        int prow = 32 * wid + 16 * mt + r + 8 * lh;
        Psm[prow][lm]      = (__bf16)p0;
        Psm[prow][16 + lm] = (__bf16)p1;
      }
    }
    __syncthreads();
    // o += P[32x32] x V_chunk[32x64]
    #pragma unroll
    for (int nt = 0; nt < 4; nt++) {
      v16bf vf;
      #pragma unroll
      for (int v = 0; v < 8; v++) {
        vf[2*v+0] = Vsm[32 * c + b_kidx(v, 0, lh)][16 * nt + lm];
        vf[2*v+1] = Vsm[32 * c + b_kidx(v, 1, lh)][16 * nt + lm];
      }
      #pragma unroll
      for (int mt = 0; mt < 2; mt++) {
        v16bf pf;
        #pragma unroll
        for (int v = 0; v < 8; v++) {
          pf[2*v+0] = Psm[32 * wid + 16 * mt + lm][a_kidx(v, 0, lh)];
          pf[2*v+1] = Psm[32 * wid + 16 * mt + lm][a_kidx(v, 1, lh)];
        }
        o[mt][nt] = wmma_bf16(pf, vf, o[mt][nt]);
      }
    }
    __syncthreads();
  }

  float* Cp = Ctx + (size_t)b * S_ * H_ + hd * DH_;
  #pragma unroll
  for (int mt = 0; mt < 2; mt++)
    #pragma unroll
    for (int r = 0; r < 8; r++) {
      float inv = 1.0f / smv[mt][r];
      int row = 32 * wid + 16 * mt + r + 8 * lh;
      #pragma unroll
      for (int nt = 0; nt < 4; nt++)
        Cp[(size_t)row * H_ + 16 * nt + lm] = o[mt][nt][r] * inv;
    }
}

// ---- Embedding gather + LayerNorm (E=128) ----------------------------------
__global__ __launch_bounds__(128) void embed_kernel(
    const int* __restrict__ ids, const float* __restrict__ wemb,
    const float* __restrict__ pemb, const float* __restrict__ temb,
    const float* __restrict__ g, const float* __restrict__ bb,
    float* __restrict__ out)
{
  __shared__ float red[128];
  int bs = blockIdx.x, s = bs % S_, t = threadIdx.x;
  float x = wemb[(size_t)ids[bs] * E_ + t] + pemb[(size_t)s * E_ + t] + temb[t];
  red[t] = x; __syncthreads();
  for (int off = 64; off > 0; off >>= 1) { if (t < off) red[t] += red[t + off]; __syncthreads(); }
  float mu = red[0] / E_; __syncthreads();
  float d = x - mu;
  red[t] = d * d; __syncthreads();
  for (int off = 64; off > 0; off >>= 1) { if (t < off) red[t] += red[t + off]; __syncthreads(); }
  float var = red[0] / E_;
  out[(size_t)bs * E_ + t] = d * rsqrtf(var + 1e-12f) * g[t] + bb[t];
}

// ---- Residual add + LayerNorm over H=768 -----------------------------------
__global__ __launch_bounds__(256) void add_ln_kernel(
    float* __restrict__ h, const float* __restrict__ a,
    const float* __restrict__ g, const float* __restrict__ bb)
{
  __shared__ float red[256];
  int row = blockIdx.x, t = threadIdx.x;
  float v[3]; float s = 0.0f;
  #pragma unroll
  for (int i = 0; i < 3; i++) {
    int c = t + 256 * i;
    v[i] = h[(size_t)row * H_ + c] + a[(size_t)row * H_ + c];
    s += v[i];
  }
  red[t] = s; __syncthreads();
  for (int off = 128; off > 0; off >>= 1) { if (t < off) red[t] += red[t + off]; __syncthreads(); }
  float mu = red[0] / H_; __syncthreads();
  s = 0.0f;
  #pragma unroll
  for (int i = 0; i < 3; i++) { float d = v[i] - mu; s += d * d; }
  red[t] = s; __syncthreads();
  for (int off = 128; off > 0; off >>= 1) { if (t < off) red[t] += red[t + off]; __syncthreads(); }
  float inv = rsqrtf(red[0] / H_ + 1e-12f);
  #pragma unroll
  for (int i = 0; i < 3; i++) {
    int c = t + 256 * i;
    h[(size_t)row * H_ + c] = (v[i] - mu) * inv * g[c] + bb[c];
  }
}

// ---- GRU scan (one block per batch, torch gate order r,z,n) ----------------
__global__ __launch_bounds__(128) void gru_kernel(
    const float* __restrict__ gx, const float* __restrict__ whh,
    const float* __restrict__ bhh, float* __restrict__ seq, int dir)
{
  __shared__ float hsm[G_];
  int b = blockIdx.x, t = threadIdx.x;
  hsm[t] = 0.0f; __syncthreads();
  for (int step = 0; step < S_; step++) {
    int s = dir ? (S_ - 1 - step) : step;
    const float* g = gx + (size_t)(b * S_ + s) * (3 * G_);
    float gh0 = bhh[t], gh1 = bhh[t + G_], gh2 = bhh[t + 2 * G_];
    for (int k = 0; k < G_; k++) {
      float hk = hsm[k];
      gh0 += whh[(size_t)t * G_ + k] * hk;
      gh1 += whh[(size_t)(t + G_) * G_ + k] * hk;
      gh2 += whh[(size_t)(t + 2 * G_) * G_ + k] * hk;
    }
    float r = 1.0f / (1.0f + __expf(-(g[t] + gh0)));
    float z = 1.0f / (1.0f + __expf(-(g[t + G_] + gh1)));
    float n = tanhf(g[t + 2 * G_] + r * gh2);
    float hold = hsm[t];
    float hnew = (1.0f - z) * n + z * hold;
    __syncthreads();
    hsm[t] = hnew;
    __syncthreads();
    seq[(size_t)(b * S_ + s) * (2 * G_) + dir * G_ + t] = hnew;
  }
}

// ---- Classifier: emissions[B*S, 9] = seq[B*S,256] x cls_w + cls_b ----------
__global__ __launch_bounds__(32) void cls_kernel(
    const float* __restrict__ seq, const float* __restrict__ w,
    const float* __restrict__ b, float* __restrict__ emis)
{
  int row = blockIdx.x, t = threadIdx.x;
  if (t < L_) {
    float acc = b[t];
    for (int k = 0; k < 2 * G_; k++) acc += seq[(size_t)row * (2 * G_) + k] * w[k * L_ + t];
    emis[(size_t)row * L_ + t] = acc;
  }
}

// ---- CRF log-likelihood per batch ------------------------------------------
__global__ __launch_bounds__(32) void crf_kernel(
    const float* __restrict__ emis, const int* __restrict__ labels,
    const int* __restrict__ mask, const float* __restrict__ trans,
    const float* __restrict__ start, const float* __restrict__ endv,
    float* __restrict__ ll)
{
  __shared__ float alpha[L_];
  __shared__ float num_s;
  int b = blockIdx.x, t = threadIdx.x;
  const float* e  = emis + (size_t)b * S_ * L_;
  const int*   tg = labels + b * S_;
  const int*   mk = mask + b * S_;
  if (t == 0) {
    int prev = tg[0];
    float sc = start[prev] + e[prev];
    for (int s = 1; s < S_; s++) {
      int ts = tg[s];
      float m = (float)mk[s];
      sc += m * (trans[prev * L_ + ts] + e[s * L_ + ts]);
      if (mk[s]) prev = ts;
    }
    num_s = sc + endv[prev];
  }
  if (t < L_) alpha[t] = start[t] + e[t];
  __syncthreads();
  for (int s = 1; s < S_; s++) {
    float nv = 0.0f;
    if (t < L_) {
      float mx = -1e30f;
      for (int i = 0; i < L_; i++) mx = fmaxf(mx, alpha[i] + trans[i * L_ + t]);
      float sum = 0.0f;
      for (int i = 0; i < L_; i++) sum += __expf(alpha[i] + trans[i * L_ + t] - mx);
      nv = mx + __logf(sum) + e[s * L_ + t];
    }
    __syncthreads();
    if (t < L_ && mk[s]) alpha[t] = nv;
    __syncthreads();
  }
  if (t == 0) {
    float mx = -1e30f;
    for (int i = 0; i < L_; i++) mx = fmaxf(mx, alpha[i] + endv[i]);
    float sum = 0.0f;
    for (int i = 0; i < L_; i++) sum += __expf(alpha[i] + endv[i] - mx);
    ll[b] = num_s - (mx + __logf(sum));
  }
}

__global__ __launch_bounds__(32) void loss_kernel(const float* __restrict__ ll,
                                                  float* __restrict__ out)
{
  if (threadIdx.x == 0) {
    float s = 0.0f;
    for (int i = 0; i < B_; i++) s += ll[i];
    out[0] = -s / (float)B_;
  }
}

// ---- Host-side pipeline -----------------------------------------------------
extern "C" void kernel_launch(void* const* d_in, const int* in_sizes, int n_in,
                              void* d_out, int out_size, void* d_ws, size_t ws_size,
                              hipStream_t stream)
{
  (void)in_sizes; (void)n_in; (void)out_size; (void)ws_size;
  const int* input_ids = (const int*)d_in[0];
  const int* attn_mask = (const int*)d_in[1];
  const int* labels    = (const int*)d_in[2];
  int pi = 3;
  const float* word_emb = (const float*)d_in[pi++];
  const float* pos_emb  = (const float*)d_in[pi++];
  const float* tok_emb  = (const float*)d_in[pi++];
  const float* ln_emb_g = (const float*)d_in[pi++];
  const float* ln_emb_b = (const float*)d_in[pi++];
  const float* proj_w   = (const float*)d_in[pi++];
  const float* proj_b   = (const float*)d_in[pi++];
  const float* wq = (const float*)d_in[pi++]; const float* bq = (const float*)d_in[pi++];
  const float* wk = (const float*)d_in[pi++]; const float* bk = (const float*)d_in[pi++];
  const float* wv = (const float*)d_in[pi++]; const float* bv = (const float*)d_in[pi++];
  const float* wo = (const float*)d_in[pi++]; const float* bo = (const float*)d_in[pi++];
  const float* ln1_g = (const float*)d_in[pi++]; const float* ln1_b = (const float*)d_in[pi++];
  const float* w1 = (const float*)d_in[pi++]; const float* b1 = (const float*)d_in[pi++];
  const float* w2 = (const float*)d_in[pi++]; const float* b2 = (const float*)d_in[pi++];
  const float* ln2_g = (const float*)d_in[pi++]; const float* ln2_b = (const float*)d_in[pi++];
  const float* gru_wih_f = (const float*)d_in[pi++]; const float* gru_whh_f = (const float*)d_in[pi++];
  const float* gru_bih_f = (const float*)d_in[pi++]; const float* gru_bhh_f = (const float*)d_in[pi++];
  const float* gru_wih_b = (const float*)d_in[pi++]; const float* gru_whh_b = (const float*)d_in[pi++];
  const float* gru_bih_b = (const float*)d_in[pi++]; const float* gru_bhh_b = (const float*)d_in[pi++];
  const float* cls_w = (const float*)d_in[pi++]; const float* cls_b = (const float*)d_in[pi++];
  const float* trans = (const float*)d_in[pi++];
  const float* startv = (const float*)d_in[pi++];
  const float* endv   = (const float*)d_in[pi++];

  float* out = (float*)d_out;
  float* ws  = (float*)d_ws;
  size_t off = 0;
  auto alloc = [&](size_t n) { float* p = ws + off; off += n; return p; };
  const int M = B_ * S_;                   // 8192 token rows
  float* xln = alloc((size_t)M * E_);
  float* h   = alloc((size_t)M * H_);
  float* qb  = alloc((size_t)M * H_);
  float* kb  = alloc((size_t)M * H_);
  float* vb  = alloc((size_t)M * H_);
  float* ctx = alloc((size_t)M * H_);
  float* tmp = alloc((size_t)M * H_);
  float* f1  = alloc((size_t)M * FFN_);
  float* gxf = alloc((size_t)M * 3 * G_);
  float* gxb = alloc((size_t)M * 3 * G_);
  float* seq = alloc((size_t)M * 2 * G_);
  float* llb = alloc(64);

  embed_kernel<<<M, 128, 0, stream>>>(input_ids, word_emb, pos_emb, tok_emb,
                                      ln_emb_g, ln_emb_b, xln);
  gemm_kernel<0,0><<<dim3(H_/128, M/128), 256, 0, stream>>>(xln, proj_w, proj_b, h, M, H_, E_);

  for (int l = 0; l < NL_; l++) {
    gemm_kernel<0,0><<<dim3(H_/128, M/128), 256, 0, stream>>>(h, wq, bq, qb, M, H_, H_);
    gemm_kernel<0,0><<<dim3(H_/128, M/128), 256, 0, stream>>>(h, wk, bk, kb, M, H_, H_);
    gemm_kernel<0,0><<<dim3(H_/128, M/128), 256, 0, stream>>>(h, wv, bv, vb, M, H_, H_);
    attn_kernel<<<B_ * NH_, 256, 0, stream>>>(qb, kb, vb, attn_mask, ctx);
    gemm_kernel<0,0><<<dim3(H_/128, M/128), 256, 0, stream>>>(ctx, wo, bo, tmp, M, H_, H_);
    add_ln_kernel<<<M, 256, 0, stream>>>(h, tmp, ln1_g, ln1_b);
    gemm_kernel<1,0><<<dim3(FFN_/128, M/128), 256, 0, stream>>>(h, w1, b1, f1, M, FFN_, H_);
    gemm_kernel<0,0><<<dim3(H_/128, M/128), 256, 0, stream>>>(f1, w2, b2, tmp, M, H_, FFN_);
    add_ln_kernel<<<M, 256, 0, stream>>>(h, tmp, ln2_g, ln2_b);
  }

  // GRU input gates: x @ W_ih^T + b_ih  (W_ih is [3G, H] torch layout)
  gemm_kernel<0,1><<<dim3(3*G_/128, M/128), 256, 0, stream>>>(h, gru_wih_f, gru_bih_f, gxf, M, 3*G_, H_);
  gemm_kernel<0,1><<<dim3(3*G_/128, M/128), 256, 0, stream>>>(h, gru_wih_b, gru_bih_b, gxb, M, 3*G_, H_);
  gru_kernel<<<B_, 128, 0, stream>>>(gxf, gru_whh_f, gru_bhh_f, seq, 0);
  gru_kernel<<<B_, 128, 0, stream>>>(gxb, gru_whh_b, gru_bhh_b, seq, 1);

  cls_kernel<<<M, 32, 0, stream>>>(seq, cls_w, cls_b, out + 1);
  crf_kernel<<<B_, 32, 0, stream>>>(out + 1, labels, attn_mask, trans, startv, endv, llb);
  loss_kernel<<<1, 32, 0, stream>>>(llb, out);
}
